// SlotAttention_48584670052369
// MI455X (gfx1250) — matrix operand order repaired and checked by reading, not compiled, not run on hardware
//
#include <hip/hip_runtime.h>
#include <hip/hip_bf16.h>
#include <math.h>

typedef __bf16 bf16;
typedef __attribute__((ext_vector_type(16))) __bf16 v16bf;
typedef __attribute__((ext_vector_type(8)))  __bf16 v8bf;
typedef __attribute__((ext_vector_type(8)))  float   v8f;

#define DEV static __device__ __forceinline__

#define HH     128
#define WWD    128
#define HWN    16384
#define CFEAT  256
#define EDIM   128
#define NSLOT  64
#define NHEAD  4
#define DHEAD  32
#define NBATCH 4
#define BHN    16
#define SCALEF 0.1767766952966369f   /* 32^-0.5 */
#define EPSV   1e-6f

/* ---------------- WMMA helpers (CDNA5 wave32, 16x16x32 bf16 -> f32) -------- */

DEV v8f wmma_bf16(v16bf a, v16bf b, v8f c) {
  return __builtin_amdgcn_wmma_f32_16x16x32_bf16(false, a, false, b, (short)0, c,
                                                 false, false);
}

DEV v16bf pack16(v8bf lo, v8bf hi) {
  v16bf r;
#pragma unroll
  for (int i = 0; i < 8; ++i) { r[i] = lo[i]; r[i + 8] = hi[i]; }
  return r;
}

/* A fragment (16xK tile, K step 32) from row-major bf16 buffer.
   Lane: m = lane&15, g = lane>>4; halves 0..7 = K g*8.., halves 8..15 = K 16+g*8.. */
DEV v16bf afrag_bf16(const bf16* base, int row0, int k0, int stride) {
  int lane = threadIdx.x & 31;
  int m = lane & 15, g = lane >> 4;
  const bf16* p = base + (row0 + m) * stride + k0 + g * 8;
  return pack16(*(const v8bf*)p, *(const v8bf*)(p + 16));
}

/* B fragment (32x16): lane = column n (lane&15), g = lane>>4; halves 0..7 = K g*8.., halves 8..15 = K 16+g*8.. */
DEV v16bf bfrag(const bf16* base, int k0, int n0, int stride) {
  int lane = threadIdx.x & 31;
  int n = lane & 15, g = lane >> 4;
  const bf16* p = base + (size_t)(k0 + g * 8) * stride + n0 + n;
  v16bf r;
#pragma unroll
  for (int i = 0; i < 8; ++i) {
    r[i] = p[(size_t)i * stride];
    r[i + 8] = p[(size_t)(16 + i) * stride];
  }
  return r;
}

/* ---------------- small prep kernels -------------------------------------- */

__global__ void k_prepw(const float* wkc, const float* wvc,
                        const float* wk, const float* wv,
                        bf16* wTk, bf16* wTv, bf16* wkb, bf16* wvb) {
  int idx = blockIdx.x * 256 + threadIdx.x;
  if (idx < CFEAT * EDIM) {               /* transpose conv weights (E,C)->(C,E) */
    int c = idx >> 7, e = idx & 127;
    wTk[idx] = (bf16)wkc[e * CFEAT + c];
    wTv[idx] = (bf16)wvc[e * CFEAT + c];
  }
  if (idx < EDIM * EDIM) { wkb[idx] = (bf16)wk[idx]; wvb[idx] = (bf16)wv[idx]; }
}

__global__ void k_pos(float* pos) {
  int idx = blockIdx.x * 256 + threadIdx.x;
  if (idx >= HWN * EDIM) return;
  int p = idx >> 7, e = idx & 127;
  int i = p >> 7, j = p & 127;
  const float twopi = 6.283185307179586f;
  float yv = (float)(i + 1) / ((float)HH + 1e-6f) * twopi;
  float xv = (float)(j + 1) / ((float)WWD + 1e-6f) * twopi;
  float base = (e < 64) ? yv : xv;
  int f = (e < 64) ? e : e - 64;
  float dim = __powf(10000.0f, (float)(f >> 1) / 32.0f);
  float v = base / dim;
  pos[idx] = (f & 1) ? __cosf(v) : __sinf(v);
}

__global__ void k_slotinit(const float* s0, float* slots) {
  int idx = blockIdx.x * 256 + threadIdx.x;
  if (idx < NBATCH * NSLOT * EDIM) slots[idx] = s0[idx & (NSLOT * EDIM - 1)];
}

__global__ void k_zero(float* p, int n) {
  int idx = blockIdx.x * 256 + threadIdx.x;
  if (idx < n) p[idx] = 0.f;
}

/* -------- fused BN+ReLU -> conv(K=256) -> (+pos) -> LN -> proj(128) -------- */
/* 64-pixel row block per workgroup; two chained WMMA GEMMs from LDS.         */

__global__ void __launch_bounds__(256) k_feat(
    const float* feats, const float* gamma, const float* beta,
    const float* mean, const float* var, const bf16* wT,
    const float* ln_g, const float* ln_b, const bf16* wproj,
    const float* bias, const float* pos, bf16* kT, bf16* vbuf, int is_k) {
  __shared__ bf16  xb[64 * CFEAT];   /* 32 KB: BN-ReLU activations / reused as LN out */
  __shared__ float fkb[64 * EDIM];   /* 32 KB: conv output rows */
  int tid = threadIdx.x;
  int b = blockIdx.x >> 8;
  int p0 = (blockIdx.x & 255) * 64;

  for (int idx = tid; idx < 64 * CFEAT; idx += 256) {
    int c = idx >> 6, p = idx & 63;
    float x = feats[((size_t)b * CFEAT + c) * HWN + p0 + p];
    float y = (x - mean[c]) * rsqrtf(var[c] + 1e-5f) * gamma[c] + beta[c];
    xb[p * CFEAT + c] = (bf16)fmaxf(y, 0.0f);
  }
  __syncthreads();

  int wave = tid >> 5, lane = tid & 31;
  int mt = wave & 3, nh = wave >> 2;
  int n_l = lane & 15, g = lane >> 4;

  /* GEMM1: (64x256)x(256x128) */
  for (int nt2 = 0; nt2 < 4; ++nt2) {
    int nt = nh * 4 + nt2;
    v8f acc = {};
#pragma unroll
    for (int kt = 0; kt < 8; ++kt) {
      v16bf a  = afrag_bf16(xb, mt * 16, kt * 32, CFEAT);
      v16bf bm = bfrag(wT, kt * 32, nt * 16, EDIM);
      acc = wmma_bf16(a, bm, acc);
    }
#pragma unroll
    for (int r = 0; r < 8; ++r)
      fkb[(mt * 16 + r + g * 8) * EDIM + nt * 16 + n_l] = acc[r];
  }
  __syncthreads();

  /* +pos (k path only) and LayerNorm, bf16 out into reused LDS */
  bf16* xn = xb;
  if (tid < 64) {
    float* row = fkb + tid * EDIM;
    if (is_k) {
      const float* pr = pos + (size_t)(p0 + tid) * EDIM;
      for (int e = 0; e < EDIM; ++e) row[e] += pr[e];
    }
    float m = 0.f;
    for (int e = 0; e < EDIM; ++e) m += row[e];
    m *= (1.0f / EDIM);
    float v = 0.f;
    for (int e = 0; e < EDIM; ++e) { float d = row[e] - m; v += d * d; }
    float inv = rsqrtf(v * (1.0f / EDIM) + 1e-5f);
    for (int e = 0; e < EDIM; ++e)
      xn[tid * EDIM + e] = (bf16)((row[e] - m) * inv * ln_g[e] + ln_b[e]);
  }
  __syncthreads();

  /* GEMM2: (64x128)x(128x128) + bias, store WMMA-friendly K/V layouts */
  for (int nt2 = 0; nt2 < 4; ++nt2) {
    int nt = nh * 4 + nt2;
    v8f acc = {};
#pragma unroll
    for (int kt = 0; kt < 4; ++kt) {
      v16bf a  = afrag_bf16(xn, mt * 16, kt * 32, EDIM);
      v16bf bm = bfrag(wproj, kt * 32, nt * 16, EDIM);
      acc = wmma_bf16(a, bm, acc);
    }
    int ng = nt * 16 + n_l;
    int h = ng >> 5, dh = ng & 31;
    float bs = bias[ng];
#pragma unroll
    for (int r = 0; r < 8; ++r) {
      int key = p0 + mt * 16 + r + g * 8;
      float val = acc[r] + bs;
      if (is_k)  /* kT: (bh, dh, key) */
        kT[(((size_t)b * NHEAD + h) * DHEAD + dh) * HWN + key] = (bf16)val;
      else       /* v:  (bh, key, dh) */
        vbuf[(((size_t)b * NHEAD + h) * HWN + key) * DHEAD + dh] = (bf16)val;
    }
  }
}

/* ---------------- per-layer q projection (small, f32 VALU) ----------------- */

__global__ void __launch_bounds__(256) k_qproj(
    const float* slots, const float* lnq_g, const float* lnq_b,
    const float* wq, const float* bq, bf16* qws) {
  __shared__ float s[NSLOT * EDIM];
  __shared__ float sn[NSLOT * EDIM];
  int tid = threadIdx.x, b = blockIdx.x;
  for (int idx = tid; idx < NSLOT * EDIM; idx += 256)
    s[idx] = slots[(size_t)b * NSLOT * EDIM + idx];
  __syncthreads();
  if (tid < NSLOT) {
    float* row = s + tid * EDIM;
    float* orow = sn + tid * EDIM;
    float m = 0.f;
    for (int e = 0; e < EDIM; ++e) m += row[e];
    m *= (1.0f / EDIM);
    float v = 0.f;
    for (int e = 0; e < EDIM; ++e) { float d = row[e] - m; v += d * d; }
    float inv = rsqrtf(v * (1.0f / EDIM) + 1e-5f);
    for (int e = 0; e < EDIM; ++e) orow[e] = (row[e] - m) * inv * lnq_g[e] + lnq_b[e];
  }
  __syncthreads();
  for (int idx = tid; idx < NSLOT * EDIM; idx += 256) {
    int sr = idx >> 7, d = idx & 127;
    float acc = bq[d];
    for (int e = 0; e < EDIM; ++e) acc += sn[sr * EDIM + e] * wq[e * EDIM + d];
    int h = d >> 5, dh = d & 31;
    qws[(((size_t)b * NHEAD + h) * NSLOT + sr) * DHEAD + dh] = (bf16)acc;
  }
}

/* -------- fused dot -> slot-axis softmax -> att@v partial accumulation ----- */
/* Grid: (bh=16) x (16 key-slices of 1024); 8 waves: mt = wave&3, half = wave>>2 */

__global__ void __launch_bounds__(256) k_attn(
    const bf16* qws, const bf16* kT, const bf16* vbuf,
    float* rowsum, float* yacc, float* attout, int write_att) {
  __shared__ float sd[NSLOT * 32];   /* dot / attex tile, f32 */
  __shared__ bf16  sdb[NSLOT * 32];  /* bf16 mirror of attex for WMMA A-frags */
  __shared__ float red[128];
  __shared__ float colmax[32], colsum[32];
  __shared__ float rloc[NSLOT];
  int tid = threadIdx.x, lane = tid & 31, wave = tid >> 5;
  int mt = wave & 3, half = wave >> 2;
  int g = lane >> 4, n_l = lane & 15;
  int bh = blockIdx.x >> 4;
  int key0 = (blockIdx.x & 15) * 1024;

  /* preload q A-fragment for this wave's 16-slot tile */
  const bf16* qp = qws + ((size_t)bh * NSLOT + mt * 16 + n_l) * DHEAD;
  v16bf qf = pack16(*(const v8bf*)(qp + g * 8), *(const v8bf*)(qp + 16 + g * 8));

  if (tid < NSLOT) rloc[tid] = 0.f;
  v8f accy = {};
  __syncthreads();

  for (int it = 0; it < 32; ++it) {
    int kb = key0 + it * 32;
    { /* dot tile: 64 slots x 32 keys, one WMMA per wave (K = DH = 32) */
      v16bf bm = bfrag(kT + (size_t)bh * DHEAD * HWN, 0, kb + half * 16, HWN);
      v8f acc = {};
      acc = wmma_bf16(qf, bm, acc);
#pragma unroll
      for (int r = 0; r < 8; ++r)
        sd[(mt * 16 + r + g * 8) * 32 + half * 16 + n_l] = acc[r] * SCALEF;
    }
    __syncthreads();
    /* softmax over the 64-slot axis, per key column */
    if (tid < 128) {
      int c = tid & 31, qr = tid >> 5;
      float mx = -1e30f;
      for (int rr = 0; rr < 16; ++rr) mx = fmaxf(mx, sd[(qr * 16 + rr) * 32 + c]);
      red[qr * 32 + c] = mx;
    }
    __syncthreads();
    if (tid < 32)
      colmax[tid] = fmaxf(fmaxf(red[tid], red[32 + tid]),
                          fmaxf(red[64 + tid], red[96 + tid]));
    __syncthreads();
    if (tid < 128) {
      int c = tid & 31, qr = tid >> 5;
      float s = 0.f, cm = colmax[c];
      for (int rr = 0; rr < 16; ++rr) s += __expf(sd[(qr * 16 + rr) * 32 + c] - cm);
      red[qr * 32 + c] = s;
    }
    __syncthreads();
    if (tid < 32) colsum[tid] = red[tid] + red[32 + tid] + red[64 + tid] + red[96 + tid];
    __syncthreads();
    for (int idx = tid; idx < 2048; idx += 256) {
      int c = idx & 31;
      float e = __expf(sd[idx] - colmax[c]) / colsum[c] + EPSV;
      sd[idx] = e;
      sdb[idx] = (bf16)e;              /* bf16 mirror for the A-fragment */
      if (write_att)  /* last layer: stream unnormalized att to d_out */
        attout[(size_t)bh * (NSLOT * HWN) + (size_t)(idx >> 5) * HWN + kb + c] = e;
    }
    __syncthreads();
    if (tid < NSLOT) {  /* per-slot key-sum for later renormalization */
      float s = 0.f;
      const float* r = sd + tid * 32;
      for (int c = 0; c < 32; ++c) s += r[c];
      rloc[tid] += s;
    }
    { /* y += attex @ v, one WMMA per wave (K = 32 keys, 16 dh columns) */
      v16bf a  = afrag_bf16(sdb, mt * 16, 0, 32);
      v16bf bm = bfrag(vbuf + (size_t)bh * HWN * DHEAD, kb, half * 16, DHEAD);
      accy = wmma_bf16(a, bm, accy);
    }
    __syncthreads();
  }
  if (tid < NSLOT) atomicAdd(&rowsum[bh * NSLOT + tid], rloc[tid]);
#pragma unroll
  for (int r = 0; r < 8; ++r)
    atomicAdd(&yacc[((size_t)bh * NSLOT + mt * 16 + r + g * 8) * DHEAD +
                    half * 16 + n_l], accy[r]);
}

/* ---------------- per-batch slot update epilogue (f32) --------------------- */

DEV void ln_inplace(float* row, const float* g, const float* b) {
  float m = 0.f;
  for (int e = 0; e < EDIM; ++e) m += row[e];
  m *= (1.0f / EDIM);
  float v = 0.f;
  for (int e = 0; e < EDIM; ++e) { float d = row[e] - m; v += d * d; }
  float inv = rsqrtf(v * (1.0f / EDIM) + 1e-5f);
  for (int e = 0; e < EDIM; ++e) row[e] = (row[e] - m) * inv * g[e] + b[e];
}

__global__ void __launch_bounds__(256) k_epilogue(
    const float* yacc, const float* rowsum,
    const float* wo, const float* bo,
    const float* pre_g, const float* pre_b,
    const float* w1, const float* b1, const float* w2, const float* b2,
    const float* post_g, const float* post_b,
    const float* mw1, const float* mb1, const float* mw2, const float* mb2,
    float* slots) {
  __shared__ float A[NSLOT * EDIM];
  __shared__ float Bf[NSLOT * EDIM];
  int tid = threadIdx.x, b = blockIdx.x;
  for (int idx = tid; idx < NSLOT * EDIM; idx += 256) {
    int s = idx >> 7, d = idx & 127;
    int rs = (b * NHEAD + (d >> 5)) * NSLOT + s;
    A[idx] = yacc[(size_t)rs * DHEAD + (d & 31)] / rowsum[rs];
  }
  __syncthreads();
  for (int idx = tid; idx < NSLOT * EDIM; idx += 256) {
    int s = idx >> 7, e = idx & 127;
    float acc = bo[e];
    for (int d = 0; d < EDIM; ++d) acc += A[s * EDIM + d] * wo[d * EDIM + e];
    Bf[idx] = acc;
  }
  __syncthreads();
  if (tid < NSLOT) ln_inplace(Bf + tid * EDIM, pre_g, pre_b);
  __syncthreads();
  for (int hb = 0; hb < 2; ++hb) {  /* z += gelu(z@w1+b1)@w2+b2, 32-row halves */
    int r0 = hb * 32;
    for (int idx = tid; idx < 32 * 256; idx += 256) {
      int s = r0 + (idx >> 8), j = idx & 255;
      float acc = b1[j];
      for (int e = 0; e < EDIM; ++e) acc += Bf[s * EDIM + e] * w1[e * 256 + j];
      A[idx] = 0.5f * acc * (1.0f + erff(acc * 0.7071067811865475f));
    }
    __syncthreads();
    for (int idx = tid; idx < 32 * EDIM; idx += 256) {
      int sl = idx >> 7, e = idx & 127;
      float acc = b2[e];
      for (int j = 0; j < 256; ++j) acc += A[sl * 256 + j] * w2[j * EDIM + e];
      Bf[(r0 + sl) * EDIM + e] += acc;
    }
    __syncthreads();
  }
  if (tid < NSLOT) ln_inplace(Bf + tid * EDIM, post_g, post_b);
  __syncthreads();
  for (int idx = tid; idx < NSLOT * EDIM; idx += 256)
    A[idx] = slots[(size_t)b * NSLOT * EDIM + idx] + Bf[idx];
  __syncthreads();
  for (int hb = 0; hb < 2; ++hb) {  /* slots += relu(slots@mw1+mb1)@mw2+mb2 */
    int r0 = hb * 32;
    for (int idx = tid; idx < 32 * 256; idx += 256) {
      int s = r0 + (idx >> 8), j = idx & 255;
      float acc = mb1[j];
      for (int e = 0; e < EDIM; ++e) acc += A[s * EDIM + e] * mw1[e * 256 + j];
      Bf[idx] = fmaxf(acc, 0.f);
    }
    __syncthreads();
    for (int idx = tid; idx < 32 * EDIM; idx += 256) {
      int sl = idx >> 7, e = idx & 127;
      float acc = mb2[e];
      for (int j = 0; j < 256; ++j) acc += Bf[sl * 256 + j] * mw2[j * EDIM + e];
      A[(r0 + sl) * EDIM + e] += acc;
    }
    __syncthreads();
  }
  for (int idx = tid; idx < NSLOT * EDIM; idx += 256)
    slots[(size_t)b * NSLOT * EDIM + idx] = A[idx];
}

/* ---------------- finalize: slots out + in-place att renormalization ------- */

__global__ void k_final(const float* slots, const float* rowsum, float* out) {
  size_t idx = (size_t)blockIdx.x * 256 + threadIdx.x;
  const size_t nsl = (size_t)NBATCH * NSLOT * EDIM;
  const size_t natt = (size_t)BHN * NSLOT * HWN;
  if (idx < nsl) out[idx] = slots[idx];
  else if (idx < nsl + natt) {
    size_t j = idx - nsl;
    out[idx] = out[idx] / rowsum[j >> 14];
  }
}

/* ---------------- host launch --------------------------------------------- */

#define OFF_POS 0u
#define OFF_WTK 8388608u
#define OFF_WTV 8454144u
#define OFF_WKB 8519680u
#define OFF_WVB 8552448u
#define OFF_KT  8585216u
#define OFF_VB  25362432u
#define OFF_Q   42139648u
#define OFF_RS  42205184u
#define OFF_Y   42209280u
#define OFF_SL  42340352u

extern "C" void kernel_launch(void* const* d_in, const int* in_sizes, int n_in,
                              void* d_out, int out_size, void* d_ws, size_t ws_size,
                              hipStream_t stream) {
  (void)in_sizes; (void)n_in; (void)out_size; (void)ws_size;
  const float* feats = (const float*)d_in[0];
  const float* bnk_g = (const float*)d_in[1];
  const float* bnk_b = (const float*)d_in[2];
  const float* bnk_m = (const float*)d_in[3];
  const float* bnk_v = (const float*)d_in[4];
  const float* bnk_w = (const float*)d_in[5];
  const float* bnv_g = (const float*)d_in[6];
  const float* bnv_b = (const float*)d_in[7];
  const float* bnv_m = (const float*)d_in[8];
  const float* bnv_v = (const float*)d_in[9];
  const float* bnv_w = (const float*)d_in[10];
  const float* lnq_g = (const float*)d_in[11];
  const float* lnq_b = (const float*)d_in[12];
  const float* lnk_g = (const float*)d_in[13];
  const float* lnk_b = (const float*)d_in[14];
  const float* lnv_g = (const float*)d_in[15];
  const float* lnv_b = (const float*)d_in[16];
  const float* wq = (const float*)d_in[17];
  const float* bq = (const float*)d_in[18];
  const float* wk = (const float*)d_in[19];
  const float* bk = (const float*)d_in[20];
  const float* wv = (const float*)d_in[21];
  const float* bv = (const float*)d_in[22];
  const float* wo = (const float*)d_in[23];
  const float* bo = (const float*)d_in[24];
  const float* pre_g = (const float*)d_in[25];
  const float* pre_b = (const float*)d_in[26];
  const float* w1 = (const float*)d_in[27];
  const float* b1 = (const float*)d_in[28];
  const float* w2 = (const float*)d_in[29];
  const float* b2 = (const float*)d_in[30];
  const float* post_g = (const float*)d_in[31];
  const float* post_b = (const float*)d_in[32];
  const float* slots0 = (const float*)d_in[33];
  const float* mw1 = (const float*)d_in[34];
  const float* mb1 = (const float*)d_in[35];
  const float* mw2 = (const float*)d_in[36];
  const float* mb2 = (const float*)d_in[37];

  char* ws = (char*)d_ws;
  float* pos   = (float*)(ws + OFF_POS);
  bf16* wTk    = (bf16*)(ws + OFF_WTK);
  bf16* wTv    = (bf16*)(ws + OFF_WTV);
  bf16* wkb    = (bf16*)(ws + OFF_WKB);
  bf16* wvb    = (bf16*)(ws + OFF_WVB);
  bf16* kT     = (bf16*)(ws + OFF_KT);
  bf16* vbuf   = (bf16*)(ws + OFF_VB);
  bf16* qws    = (bf16*)(ws + OFF_Q);
  float* rsum  = (float*)(ws + OFF_RS);
  float* yacc  = (float*)(ws + OFF_Y);
  float* slots = (float*)(ws + OFF_SL);
  float* outp  = (float*)d_out;
  float* attout = outp + NBATCH * NSLOT * EDIM;

  k_prepw<<<128, 256, 0, stream>>>(bnk_w, bnv_w, wk, wv, wTk, wTv, wkb, wvb);
  k_pos<<<8192, 256, 0, stream>>>(pos);
  /* K and V are layer-invariant: compute once */
  k_feat<<<1024, 256, 0, stream>>>(feats, bnk_g, bnk_b, bnk_m, bnk_v, wTk,
                                   lnk_g, lnk_b, wkb, bk, pos, kT, vbuf, 1);
  k_feat<<<1024, 256, 0, stream>>>(feats, bnv_g, bnv_b, bnv_m, bnv_v, wTv,
                                   lnv_g, lnv_b, wvb, bv, pos, kT, vbuf, 0);
  k_slotinit<<<128, 256, 0, stream>>>(slots0, slots);

  for (int layer = 0; layer < 3; ++layer) {
    k_zero<<<132, 256, 0, stream>>>(rsum, 1024 + BHN * NSLOT * DHEAD);
    k_qproj<<<NBATCH, 256, 0, stream>>>(slots, lnq_g, lnq_b, wq, bq, qws);
    k_attn<<<256, 256, 0, stream>>>(qws, kT, vbuf, rsum, yacc, attout,
                                    layer == 2 ? 1 : 0);
    k_epilogue<<<NBATCH, 256, 0, stream>>>(yacc, rsum, wo, bo, pre_g, pre_b,
                                           w1, b1, w2, b2, post_g, post_b,
                                           mw1, mb1, mw2, mb2, slots);
  }
  k_final<<<65665, 256, 0, stream>>>(slots, rsum, outp);
}
